// EndToEndModel_86552180949694
// MI455X (gfx1250) — compile-verified, matrix operand order checked by reading, not run on hardware
//
#include <hip/hip_runtime.h>
#include <hip/hip_bf16.h>

// ---------------------------------------------------------------------------
// Model dims (match reference): B=32 S=128 V=50265 H=768 L=4 NH=12 DH=64
// FF=3072 T=3 R=16, LORA_SCALE = 2.0
// ---------------------------------------------------------------------------
constexpr int Bc = 32, Sc = 128, Hc = 768, Lc = 4, NHc = 12, DHc = 64;
constexpr int FFc = 3072, Tc = 3, Rc = 16;

typedef __attribute__((ext_vector_type(16))) __bf16 v16bf;
typedef __attribute__((ext_vector_type(8)))  float  v8f;

union Frag { unsigned int u[8]; v16bf v; };

__device__ __forceinline__ unsigned short f2bf(float f) {
  unsigned int u = __float_as_uint(f);
  unsigned int r = (u + 0x7FFFu + ((u >> 16) & 1u)) >> 16;  // RNE
  return (unsigned short)r;
}

__device__ __forceinline__ float gelu_tanh(float x) {
  return 0.5f * x * (1.0f + tanhf(0.7978845608028654f * (x + 0.044715f * x * x * x)));
}

// low 32 bits of the flat address of an LDS object == LDS byte offset
__device__ __forceinline__ unsigned lds_off(const void* p) {
  return (unsigned)(unsigned long long)(uintptr_t)p;
}

// async global->LDS copy of 16 bytes per lane (CDNA5, ASYNCcnt-tracked)
__device__ __forceinline__ void async_copy16(unsigned lofs, const void* gptr) {
  unsigned long long ga = (unsigned long long)(uintptr_t)gptr;
  asm volatile("global_load_async_to_lds_b128 %0, %1, off"
               :: "v"(lofs), "v"(ga) : "memory");
}
__device__ __forceinline__ void async_wait0() {
  asm volatile("s_wait_asynccnt 0x0" ::: "memory");
}
__device__ __forceinline__ void async_wait4() {
  asm volatile("s_wait_asynccnt 0x4" ::: "memory");
}

// ---------------------------------------------------------------------------
// Weight convert+transpose: W[K,N] f32 -> Wt[N,K] bf16.  Grid (K/64, N/64).
// ---------------------------------------------------------------------------
__global__ __launch_bounds__(256)
void conv_w_kernel(const float* __restrict__ W, unsigned short* __restrict__ Wt,
                   int K, int N) {
  __shared__ unsigned short t[64][65];
  int k0 = blockIdx.x * 64;
  int n0 = blockIdx.y * 64;
  int tid = threadIdx.x;
#pragma unroll
  for (int i = 0; i < 16; ++i) {
    int idx = tid + i * 256;
    int kk = idx >> 6, nn = idx & 63;
    t[nn][kk] = f2bf(W[(size_t)(k0 + kk) * N + n0 + nn]);
  }
  __syncthreads();
#pragma unroll
  for (int i = 0; i < 16; ++i) {
    int idx = tid + i * 256;
    int nn = idx >> 6, kk = idx & 63;
    Wt[(size_t)(n0 + nn) * K + k0 + kk] = t[nn][kk];
  }
}

// ---------------------------------------------------------------------------
// GEMM tile stage: copy A[128x32] and Bt[128x32] bf16 tiles into LDS via
// async global->LDS. 4 async instructions per thread (per wave) per tile.
// ---------------------------------------------------------------------------
__device__ __forceinline__ void stage_tile(const unsigned short* __restrict__ A,
                                           const unsigned short* __restrict__ Bt,
                                           int rowBase, int colBase, int K, int k0,
                                           unsigned ldsA, unsigned ldsB, int tid) {
  constexpr int ST = 40;
#pragma unroll
  for (int i = 0; i < 2; ++i) {
    int idx = tid + i * 256;
    int r = idx >> 2, c8 = (idx & 3) << 3;
    async_copy16(ldsA + (unsigned)(r * ST + c8) * 2,
                 A + (size_t)(rowBase + r) * K + k0 + c8);
    async_copy16(ldsB + (unsigned)(r * ST + c8) * 2,
                 Bt + (size_t)(colBase + r) * K + k0 + c8);
  }
}

// ---------------------------------------------------------------------------
// C = act(A @ Bt^T + bias); A [M,K] bf16, Bt [N,K] bf16 (pre-transposed).
// ACT: 0=none 1=gelu 2=relu.  OUTBF: store bf16 instead of fp32.
// Block: 256 thr = 8 waves (4 M x 2 N); block tile 128x128; wave tile 32x64.
// Double-buffered LDS, async copies overlapped with WMMA compute.
// Requires K%32==0, N%128==0; M row-guarded at store (out-of-tile A rows may
// be garbage -> affect only masked output rows).
// ---------------------------------------------------------------------------
template <int ACT, bool OUTBF>
__global__ __launch_bounds__(256)
void gemm_bf16_t(const unsigned short* __restrict__ A,
                 const unsigned short* __restrict__ Bt,
                 const float* __restrict__ bias,
                 float* __restrict__ Cf, unsigned short* __restrict__ Cbf,
                 int M, int N, int K) {
  constexpr int ST = 40;  // LDS row stride (bf16): 80 B -> 16B-aligned chunks
  __shared__ __align__(16) unsigned short As[2][128 * ST];
  __shared__ __align__(16) unsigned short Bs[2][128 * ST];

  const int tid    = threadIdx.x;
  const int lane   = tid & 31;
  const int wave   = tid >> 5;
  const int hlf    = (lane >> 4) & 1;
  const int lane16 = lane & 15;
  const int waveM  = wave & 3;   // 4 waves x 32 rows = 128
  const int waveN  = wave >> 2;  // 2 waves x 64 cols = 128
  const int rowBase = blockIdx.y * 128;
  const int colBase = blockIdx.x * 128;

  const unsigned ldsA0 = lds_off(&As[0][0]), ldsA1 = lds_off(&As[1][0]);
  const unsigned ldsB0 = lds_off(&Bs[0][0]), ldsB1 = lds_off(&Bs[1][0]);

  v8f acc[2][4] = {};
  const int nk = K >> 5;

  stage_tile(A, Bt, rowBase, colBase, K, 0, ldsA0, ldsB0, tid);

  for (int it = 0; it < nk; ++it) {
    const int cur = it & 1;
    if (it + 1 < nk) {
      // prefetch next tile into the other buffer, then drain only the
      // current tile's 4 async ops (in-order completion).
      stage_tile(A, Bt, rowBase, colBase, K, (it + 1) << 5,
                 cur ? ldsA0 : ldsA1, cur ? ldsB0 : ldsB1, tid);
      async_wait4();
    } else {
      async_wait0();
    }
    __syncthreads();

    const unsigned short* as = &As[cur][0];
    const unsigned short* bs = &Bs[cur][0];
    Frag af[2], bfr[4];
#pragma unroll
    for (int mi = 0; mi < 2; ++mi) {
      int m = waveM * 32 + mi * 16 + lane16;
#pragma unroll
      for (int j = 0; j < 8; ++j) {
        int kb = 2 * j + ((j >= 4) ? 8 : 0) + 8 * hlf;
        af[mi].u[j] = *(const unsigned int*)&as[m * ST + kb];
      }
    }
#pragma unroll
    for (int ni = 0; ni < 4; ++ni) {
      int n = waveN * 64 + ni * 16 + lane16;
#pragma unroll
      for (int j = 0; j < 8; ++j) {
        int kb = 2 * j + ((j >= 4) ? 8 : 0) + 8 * hlf;
        bfr[ni].u[j] = *(const unsigned int*)&bs[n * ST + kb];
      }
    }
#pragma unroll
    for (int mi = 0; mi < 2; ++mi)
#pragma unroll
      for (int ni = 0; ni < 4; ++ni)
        acc[mi][ni] = __builtin_amdgcn_wmma_f32_16x16x32_bf16(
            false, af[mi].v, false, bfr[ni].v, (short)0, acc[mi][ni], false, false);
    __syncthreads();  // all waves done reading before this buffer is refilled
  }

  // ---- branch-free epilogue
#pragma unroll
  for (int mi = 0; mi < 2; ++mi) {
#pragma unroll
    for (int ni = 0; ni < 4; ++ni) {
      int gc = colBase + waveN * 64 + ni * 16 + lane16;
      float bsv = bias[gc];
#pragma unroll
      for (int r = 0; r < 8; ++r) {
        int gr = rowBase + waveM * 32 + mi * 16 + r + 8 * hlf;
        if (gr < M) {
          float v = acc[mi][ni][r] + bsv;
          if constexpr (ACT == 1) v = gelu_tanh(v);
          if constexpr (ACT == 2) v = fmaxf(v, 0.0f);
          if constexpr (OUTBF) Cbf[(size_t)gr * N + gc] = f2bf(v);
          else                 Cf[(size_t)gr * N + gc] = v;
        }
      }
    }
  }
}

// ---------------------------------------------------------------------------
// out = LN(in (+ resid)) * g + b ; also emits bf16 copy for downstream GEMMs.
// ---------------------------------------------------------------------------
__global__ __launch_bounds__(256)
void ln_kernel(const float* __restrict__ in, const float* __restrict__ resid,
               const float* __restrict__ g, const float* __restrict__ bta,
               float* __restrict__ out, unsigned short* __restrict__ outbf) {
  __shared__ float red[256];
  const int row = blockIdx.x;
  const int tid = threadIdx.x;
  float vals[3];
  float s = 0.f;
#pragma unroll
  for (int i = 0; i < 3; ++i) {
    int h = tid + i * 256;
    float v = in[(size_t)row * Hc + h];
    if (resid) v += resid[(size_t)row * Hc + h];
    vals[i] = v;
    s += v;
  }
  red[tid] = s; __syncthreads();
  for (int off = 128; off > 0; off >>= 1) {
    if (tid < off) red[tid] += red[tid + off];
    __syncthreads();
  }
  float mean = red[0] * (1.0f / Hc);
  __syncthreads();
  float vs = 0.f;
#pragma unroll
  for (int i = 0; i < 3; ++i) { float d = vals[i] - mean; vs += d * d; }
  red[tid] = vs; __syncthreads();
  for (int off = 128; off > 0; off >>= 1) {
    if (tid < off) red[tid] += red[tid + off];
    __syncthreads();
  }
  float rstd = rsqrtf(red[0] * (1.0f / Hc) + 1e-5f);
#pragma unroll
  for (int i = 0; i < 3; ++i) {
    int h = tid + i * 256;
    float o = (vals[i] - mean) * rstd * g[h] + bta[h];
    out[(size_t)row * Hc + h] = o;
    if (outbf) outbf[(size_t)row * Hc + h] = f2bf(o);
  }
}

__global__ __launch_bounds__(256)
void embed_kernel(const int* __restrict__ ids, const float* __restrict__ wemb,
                  const float* __restrict__ pemb, float* __restrict__ X) {
  int token = blockIdx.x;
  int s = token & (Sc - 1);
  int id = ids[token];
#pragma unroll
  for (int i = 0; i < 3; ++i) {
    int h = threadIdx.x + i * 256;
    X[(size_t)token * Hc + h] = wemb[(size_t)id * Hc + h] + pemb[(size_t)s * Hc + h];
  }
}

// ---------------------------------------------------------------------------
// Attention: block per (b, head, query); 128 threads = one per key.
// ctx emitted as bf16 (only consumed by the o-projection GEMM).
// ---------------------------------------------------------------------------
__global__ __launch_bounds__(128)
void attn_kernel(const float* __restrict__ Q, const float* __restrict__ Kb,
                 const float* __restrict__ Vb, const int* __restrict__ mask,
                 unsigned short* __restrict__ ctx) {
  __shared__ float qs[DHc];
  __shared__ float p[Sc];
  __shared__ float red[Sc];
  int blk = blockIdx.x;                 // b*NH*S + h*S + qi
  int qi = blk & (Sc - 1);
  int hh = (blk >> 7) % NHc;
  int b  = blk / (NHc * Sc);
  int tid = threadIdx.x;
  const size_t qoff = ((size_t)(b * Sc + qi) * Hc) + hh * DHc;
  if (tid < DHc) qs[tid] = Q[qoff + tid];
  __syncthreads();

  const float* krow = Kb + ((size_t)(b * Sc + tid) * Hc) + hh * DHc;
  float sc = 0.f;
#pragma unroll 8
  for (int d = 0; d < DHc; ++d) sc += qs[d] * krow[d];
  sc *= 0.125f;
  sc += (mask[b * Sc + tid] != 0) ? 0.f : -1e9f;

  red[tid] = sc; __syncthreads();
  for (int off = 64; off > 0; off >>= 1) {
    if (tid < off) red[tid] = fmaxf(red[tid], red[tid + off]);
    __syncthreads();
  }
  float mx = red[0]; __syncthreads();
  float e = __expf(sc - mx);
  red[tid] = e; __syncthreads();
  for (int off = 64; off > 0; off >>= 1) {
    if (tid < off) red[tid] += red[tid + off];
    __syncthreads();
  }
  float inv = 1.0f / red[0];
  p[tid] = e * inv;
  __syncthreads();

  if (tid < DHc) {
    float acc = 0.f;
    const float* vcol = Vb + ((size_t)(b * Sc) * Hc) + hh * DHc + tid;
    for (int k = 0; k < Sc; ++k) acc += p[k] * vcol[(size_t)k * Hc];
    ctx[qoff + tid] = f2bf(acc);
  }
}

// ---------------------------------------------------------------------------
// LoRA rank-16: outq[token] += 2.0 * (x A^T) B^T, adapters gathered by pred[b]
// ---------------------------------------------------------------------------
__global__ __launch_bounds__(256)
void lora_kernel(const float* __restrict__ X, const float* __restrict__ Am,
                 const float* __restrict__ Bm, const int* __restrict__ pred,
                 float* __restrict__ outq, int l) {
  __shared__ float xs[Hc];
  __shared__ float tt[Rc];
  __shared__ float red[256];
  int token = blockIdx.x;
  int b = token >> 7;
  int tid = threadIdx.x;
#pragma unroll
  for (int i = 0; i < 3; ++i)
    xs[tid + i * 256] = X[(size_t)token * Hc + tid + i * 256];
  __syncthreads();
  int p = pred[b];
  const float* Ab = Am + ((size_t)p * Lc + l) * Rc * Hc;
  const float* Bb = Bm + ((size_t)p * Lc + l) * Hc * Rc;
  int r = tid >> 4;
  int u = tid & 15;
  float s = 0.f;
  for (int h = u; h < Hc; h += 16) s += xs[h] * Ab[(size_t)r * Hc + h];
  red[tid] = s; __syncthreads();
#pragma unroll
  for (int off = 8; off > 0; off >>= 1) {
    if (u < off) red[tid] += red[tid + off];
    __syncthreads();
  }
  if (u == 0) tt[r] = red[tid];
  __syncthreads();
#pragma unroll
  for (int i = 0; i < 3; ++i) {
    int h = tid + i * 256;
    float acc = 0.f;
#pragma unroll
    for (int rr = 0; rr < Rc; ++rr) acc += tt[rr] * Bb[(size_t)h * Rc + rr];
    outq[(size_t)token * Hc + h] += 2.0f * acc;
  }
}

__global__ __launch_bounds__(256)
void gather_cls(const float* __restrict__ X, float* __restrict__ cls,
                unsigned short* __restrict__ clsbf) {
  int b = blockIdx.x;
#pragma unroll
  for (int i = 0; i < 3; ++i) {
    int h = threadIdx.x + i * 256;
    float v = X[(size_t)b * Sc * Hc + h];
    cls[(size_t)b * Hc + h] = v;
    clsbf[(size_t)b * Hc + h] = f2bf(v);
  }
}

__global__ __launch_bounds__(256)
void router_out_kernel(const float* __restrict__ h1, const float* __restrict__ W2,
                       const float* __restrict__ b2, const float* __restrict__ temp,
                       float* __restrict__ logits_out, int* __restrict__ pred) {
  __shared__ float red[256];
  __shared__ float lg[Tc];
  int b = blockIdx.x, tid = threadIdx.x;
  for (int t = 0; t < Tc; ++t) {
    float s = 0.f;
    for (int h = tid; h < Hc; h += 256) s += h1[(size_t)b * Hc + h] * W2[(size_t)h * Tc + t];
    red[tid] = s; __syncthreads();
    for (int off = 128; off > 0; off >>= 1) {
      if (tid < off) red[tid] += red[tid + off];
      __syncthreads();
    }
    if (tid == 0) {
      float tv = fmaxf(temp[0], 0.05f);
      float v = (red[0] + b2[t]) / tv;
      lg[t] = v;
      logits_out[b * Tc + t] = v;
    }
    __syncthreads();
  }
  if (tid == 0) {
    int best = 0; float bv = lg[0];
    for (int t = 1; t < Tc; ++t) if (lg[t] > bv) { bv = lg[t]; best = t; }
    pred[b] = best;
  }
}

__global__ __launch_bounds__(256)
void cls_dense_kernel(const float* __restrict__ cls, const float* __restrict__ Wd,
                      const float* __restrict__ bd, const int* __restrict__ pred,
                      float* __restrict__ out) {
  __shared__ float xs[Hc];
  int b = blockIdx.y;
  int o = blockIdx.x * 256 + threadIdx.x;
#pragma unroll
  for (int i = 0; i < 3; ++i)
    xs[threadIdx.x + i * 256] = cls[(size_t)b * Hc + threadIdx.x + i * 256];
  __syncthreads();
  int p = pred[b];
  const float* Wm = Wd + (size_t)p * Hc * Hc;
  float acc = 0.f;
  for (int h = 0; h < Hc; ++h) acc += xs[h] * Wm[(size_t)h * Hc + o];
  out[(size_t)b * Hc + o] = tanhf(acc + bd[p * Hc + o]);
}

__global__ __launch_bounds__(256)
void cls_out_kernel(const float* __restrict__ h2, const float* __restrict__ Wo,
                    const float* __restrict__ bo, const int* __restrict__ pred,
                    float* __restrict__ out) {
  __shared__ float red[256];
  int b = blockIdx.x, tid = threadIdx.x;
  int p = pred[b];
  for (int j = 0; j < 2; ++j) {
    float s = 0.f;
    for (int h = tid; h < Hc; h += 256)
      s += h2[(size_t)b * Hc + h] * Wo[(size_t)p * Hc * 2 + h * 2 + j];
    red[tid] = s; __syncthreads();
    for (int off = 128; off > 0; off >>= 1) {
      if (tid < off) red[tid] += red[tid + off];
      __syncthreads();
    }
    if (tid == 0) out[b * 2 + j] = red[0] + bo[p * 2 + j];
    __syncthreads();
  }
}

// ---------------------------------------------------------------------------
// Host orchestration
// ---------------------------------------------------------------------------
extern "C" void kernel_launch(void* const* d_in, const int* in_sizes, int n_in,
                              void* d_out, int out_size, void* d_ws, size_t ws_size,
                              hipStream_t stream) {
  (void)in_sizes; (void)n_in; (void)out_size; (void)ws_size;

  const int*   ids   = (const int*)d_in[0];
  const int*   mask  = (const int*)d_in[1];
  /* d_in[2] = task_id (unused in forward) */
  const float* wemb  = (const float*)d_in[3];
  const float* pemb  = (const float*)d_in[4];
  const float* eg    = (const float*)d_in[5];
  const float* ebi   = (const float*)d_in[6];
  const float* Wq    = (const float*)d_in[7];  const float* bq  = (const float*)d_in[8];
  const float* Wk    = (const float*)d_in[9];  const float* bk  = (const float*)d_in[10];
  const float* Wv    = (const float*)d_in[11]; const float* bv  = (const float*)d_in[12];
  const float* Wo    = (const float*)d_in[13]; const float* bo  = (const float*)d_in[14];
  const float* ln1g  = (const float*)d_in[15]; const float* ln1b = (const float*)d_in[16];
  const float* W1    = (const float*)d_in[17]; const float* b1  = (const float*)d_in[18];
  const float* W2    = (const float*)d_in[19]; const float* b2  = (const float*)d_in[20];
  const float* ln2g  = (const float*)d_in[21]; const float* ln2b = (const float*)d_in[22];
  const float* Aq    = (const float*)d_in[23]; const float* Bq  = (const float*)d_in[24];
  const float* Av    = (const float*)d_in[25]; const float* Bv  = (const float*)d_in[26];
  const float* cdW   = (const float*)d_in[27]; const float* cdb = (const float*)d_in[28];
  const float* coW   = (const float*)d_in[29]; const float* cob = (const float*)d_in[30];
  const float* rf1W  = (const float*)d_in[31]; const float* rf1b = (const float*)d_in[32];
  const float* rf2W  = (const float*)d_in[33]; const float* rf2b = (const float*)d_in[34];
  const float* rtemp = (const float*)d_in[35];

  float* out = (float*)d_out;

  const size_t BSH = (size_t)Bc * Sc * Hc;     // 3,145,728
  const size_t BSF = (size_t)Bc * Sc * FFc;    // 12,582,912
  const size_t WHH = (size_t)Hc * Hc;          // 589,824
  const size_t WLAYER = 4 * WHH + 2 * (size_t)Hc * FFc;  // per-layer bf16 weights

  float* ws   = (float*)d_ws;
  float* X    = ws;
  float* TMP  = ws + 1 * BSH;
  float* Qb   = ws + 2 * BSH;
  float* Kbuf = ws + 3 * BSH;
  float* Vbuf = ws + 4 * BSH;
  unsigned short* u16b = (unsigned short*)(ws + 5 * BSH);
  unsigned short* Xbf  = u16b;                 // [B*S, H]
  unsigned short* CTbf = Xbf + BSH;            // [B*S, H]
  unsigned short* FFbf = CTbf + BSH;           // [B*S, FF]
  unsigned short* WtL  = FFbf + BSF;           // per-layer transposed weights
  unsigned short* Rf1t = WtL + WLAYER;         // router fc1 transposed
  unsigned short* CLSbf = Rf1t + WHH;          // [128, H] (padded to GEMM tile)
  float* CLS  = (float*)(CLSbf + (size_t)128 * Hc);
  float* H1   = CLS + (size_t)Bc * Hc;
  int*   PRED = (int*)(H1 + (size_t)Bc * Hc);

  const int Mtok = Bc * Sc;  // 4096
  dim3 gProj(Hc / 128, Mtok / 128);    // 6 x 32
  dim3 gFF1(FFc / 128, Mtok / 128);    // 24 x 32
  dim3 gFF2(Hc / 128, Mtok / 128);
  dim3 gRout(Hc / 128, 1);             // M = 32 (row-guarded)

  auto convW = [&](const float* Wsrc, unsigned short* Wdst, int K, int N) {
    conv_w_kernel<<<dim3(K / 64, N / 64), 256, 0, stream>>>(Wsrc, Wdst, K, N);
  };

  unsigned short* Wqt = WtL + 0 * WHH;
  unsigned short* Wkt = WtL + 1 * WHH;
  unsigned short* Wvt = WtL + 2 * WHH;
  unsigned short* Wot = WtL + 3 * WHH;
  unsigned short* W1t = WtL + 4 * WHH;                       // [FF, H] layout
  unsigned short* W2t = WtL + 4 * WHH + (size_t)Hc * FFc;    // [H, FF] layout

  auto encoder = [&](bool use_lora) {
    embed_kernel<<<Mtok, 256, 0, stream>>>(ids, wemb, pemb, TMP);
    ln_kernel<<<Mtok, 256, 0, stream>>>(TMP, nullptr, eg, ebi, X, Xbf);
    for (int l = 0; l < Lc; ++l) {
      const size_t hh = (size_t)l * WHH;
      convW(Wq + hh, Wqt, Hc, Hc);
      convW(Wk + hh, Wkt, Hc, Hc);
      convW(Wv + hh, Wvt, Hc, Hc);
      convW(Wo + hh, Wot, Hc, Hc);
      convW(W1 + (size_t)l * Hc * FFc, W1t, Hc, FFc);
      convW(W2 + (size_t)l * FFc * Hc, W2t, FFc, Hc);

      gemm_bf16_t<0, false><<<gProj, 256, 0, stream>>>(Xbf, Wqt, bq + l * Hc, Qb, nullptr, Mtok, Hc, Hc);
      gemm_bf16_t<0, false><<<gProj, 256, 0, stream>>>(Xbf, Wkt, bk + l * Hc, Kbuf, nullptr, Mtok, Hc, Hc);
      gemm_bf16_t<0, false><<<gProj, 256, 0, stream>>>(Xbf, Wvt, bv + l * Hc, Vbuf, nullptr, Mtok, Hc, Hc);
      if (use_lora) {
        lora_kernel<<<Mtok, 256, 0, stream>>>(X, Aq, Bq, PRED, Qb, l);
        lora_kernel<<<Mtok, 256, 0, stream>>>(X, Av, Bv, PRED, Vbuf, l);
      }
      attn_kernel<<<Bc * NHc * Sc, 128, 0, stream>>>(Qb, Kbuf, Vbuf, mask, CTbf);
      gemm_bf16_t<0, false><<<gProj, 256, 0, stream>>>(CTbf, Wot, bo + l * Hc, TMP, nullptr, Mtok, Hc, Hc);
      ln_kernel<<<Mtok, 256, 0, stream>>>(TMP, X, ln1g + l * Hc, ln1b + l * Hc, X, Xbf);
      gemm_bf16_t<1, true><<<gFF1, 256, 0, stream>>>(Xbf, W1t, b1 + l * FFc, nullptr, FFbf, Mtok, FFc, Hc);
      gemm_bf16_t<0, false><<<gFF2, 256, 0, stream>>>(FFbf, W2t, b2 + l * Hc, TMP, nullptr, Mtok, Hc, FFc);
      ln_kernel<<<Mtok, 256, 0, stream>>>(TMP, X, ln2g + l * Hc, ln2b + l * Hc, X, Xbf);
    }
  };

  // router fc1 weights (bf16, transposed) once
  convW(rf1W, Rf1t, Hc, Hc);

  // Pass 1: shared frozen encoder -> router -> pred
  encoder(false);
  gather_cls<<<Bc, 256, 0, stream>>>(X, CLS, CLSbf);
  gemm_bf16_t<2, false><<<gRout, 256, 0, stream>>>(CLSbf, Rf1t, rf1b, H1, nullptr, Bc, Hc, Hc);
  router_out_kernel<<<Bc, 256, 0, stream>>>(H1, rf2W, rf2b, rtemp, out, PRED);

  // Pass 2: encoder with per-sample LoRA adapters -> per-sample heads
  encoder(true);
  gather_cls<<<Bc, 256, 0, stream>>>(X, CLS, CLSbf);
  cls_dense_kernel<<<dim3(Hc / 256, Bc), 256, 0, stream>>>(CLS, cdW, cdb, PRED, H1);
  cls_out_kernel<<<Bc, 256, 0, stream>>>(H1, coW, cob, PRED, out + Bc * Tc);
}
